// LinearAttentionTransformer_68401649156639
// MI455X (gfx1250) — matrix operand order repaired and checked by reading, not compiled, not run on hardware
//
#include <hip/hip_runtime.h>

// ---------------------------------------------------------------------------
// Causal linear attention transformer block, chunk-parallel, CDNA5 WMMA + TDM.
// B=2, N=2048, DIM=512, H=8, DH=64.  Chunk size = 64.
// ---------------------------------------------------------------------------

typedef __attribute__((ext_vector_type(16))) __bf16 bf16x16;
typedef __attribute__((ext_vector_type(8)))  float  f32x8;
typedef __attribute__((ext_vector_type(4)))  unsigned int u32x4;
typedef __attribute__((ext_vector_type(4)))  int    i32x4;
typedef __attribute__((ext_vector_type(8)))  int    i32x8;
typedef unsigned int uint32;

#define EPS 1e-3f

// ---- bf16 <-> f32, native converts -----------------------------------------
__device__ __forceinline__ __bf16 f2bf(float f) { return (__bf16)f; }
__device__ __forceinline__ float  bf2f(__bf16 h) { return (float)h; }

// ---- WMMA wrapper -----------------------------------------------------------
__device__ __forceinline__ f32x8 wmma_bf16(bf16x16 a, bf16x16 b, f32x8 c) {
  return __builtin_amdgcn_wmma_f32_16x16x32_bf16(
      /*neg_a=*/false, a, /*neg_b=*/false, b,
      /*c_mod=*/(short)0, c, /*reuse_a=*/false, /*reuse_b=*/false);
}

// ---- Tensor Data Mover: 2D bf16 tile, global (row stride) -> LDS (dense) ---
// D# per CDNA5 ISA ch.8: group0 = {flags, lds_addr, global_addr, type=2},
// group1 = {data_size=2B, tensor dims, tile dims, dim0 stride}, groups 2/3 = 0.
__device__ __forceinline__ void tdm_load_2d_bf16(uint32 ldsOff, const void* gptr,
                                                 int cols, int rows, int srcStride) {
  unsigned long long ga = (unsigned long long)(uintptr_t)gptr;
  u32x4 g0;
  g0[0] = 1u;                                   // count=1, user descriptor
  g0[1] = ldsOff;                               // lds_addr (bytes)
  g0[2] = (uint32)(ga & 0xFFFFFFFFu);           // global_addr[31:0]
  g0[3] = (uint32)((ga >> 32) & 0x01FFFFFFu)    // global_addr[56:32]
        | (2u << 30);                           // type = 2 ("image")
  i32x8 g1;
  g1[0] = (1 << 16);                            // data_size = 1 (2 bytes)
  g1[1] = (cols & 0xFFFF) << 16;                // tensor_dim0[15:0]
  g1[2] = ((cols >> 16) & 0xFFFF)               // tensor_dim0[31:16]
        | ((rows & 0xFFFF) << 16);              // tensor_dim1[15:0]
  g1[3] = ((rows >> 16) & 0xFFFF)               // tensor_dim1[31:16]
        | ((cols & 0xFFFF) << 16);              // tile_dim0
  g1[4] = (rows & 0xFFFF);                      // tile_dim1 (tile_dim2 = 0)
  g1[5] = srcStride;                            // tensor_dim0_stride[31:0]
  g1[6] = 0;                                    // stride hi bits
  g1[7] = 0;
  i32x4 gz = {0, 0, 0, 0};
#if defined(__clang_major__) && __clang_major__ >= 23
  i32x8 gz8 = {0, 0, 0, 0, 0, 0, 0, 0};
  __builtin_amdgcn_tensor_load_to_lds(g0, g1, gz, gz, gz8, 0);
#else
  __builtin_amdgcn_tensor_load_to_lds(g0, g1, gz, gz, 0);
#endif
}
__device__ __forceinline__ void tdm_wait0() {
  __builtin_amdgcn_s_wait_tensorcnt(0);
}
__device__ __forceinline__ uint32 lds_off(const void* p) {
  // LDS aperture: ADDR[31:0] is the LDS byte offset (ISA 10.2)
  return (uint32)(uintptr_t)p;
}

// ---- fragment loaders from LDS (row-major source, leading dim ld) ----------
// A operand 16x32: lanes 0-15 rows 0-15 with K in {0..7,16..23};
//                  lanes 16-31 rows 0-15 with K in {8..15,24..31}
__device__ __forceinline__ bf16x16 ldsA(const __bf16* p, int ld) {
  int lane = threadIdx.x & 31;
  int row  = lane & 15;
  int kof  = (lane >> 4) * 8;
  bf16x16 a;
#pragma unroll
  for (int j = 0; j < 8; ++j) {
    int k = ((j < 4) ? (2 * j) : (16 + 2 * (j - 4))) + kof;
    a[2 * j]     = p[row * ld + k];
    a[2 * j + 1] = p[row * ld + k + 1];
  }
  return a;
}
// A operand, logically transposed: A[row][k] = src[k][row]; p -> src[k0][row0]
__device__ __forceinline__ bf16x16 ldsAT(const __bf16* p, int ld) {
  int lane = threadIdx.x & 31;
  int row  = lane & 15;
  int kof  = (lane >> 4) * 8;
  bf16x16 a;
#pragma unroll
  for (int j = 0; j < 8; ++j) {
    int k = ((j < 4) ? (2 * j) : (16 + 2 * (j - 4))) + kof;
    a[2 * j]     = p[k * ld + row];
    a[2 * j + 1] = p[(k + 1) * ld + row];
  }
  return a;
}
// B operand 32x16: lane%16 = N column, elements = K in {0..7,16..23} (lanes 0-15) / {8..15,24..31} (lanes 16-31); p -> src[k0][n0]
__device__ __forceinline__ bf16x16 ldsB(const __bf16* p, int ld) {
  int lane = threadIdx.x & 31;
  int col  = lane & 15;
  int kof  = (lane >> 4) * 8;
  bf16x16 b;
#pragma unroll
  for (int e = 0; e < 16; ++e) b[e] = p[(((e < 8) ? e : (8 + e)) + kof) * ld + col];
  return b;
}
// B operand, logically transposed: B[k][n] = src[n][k]; p -> src[n0][k0]
__device__ __forceinline__ bf16x16 ldsBT(const __bf16* p, int ld) {
  int lane = threadIdx.x & 31;
  int col  = lane & 15;
  int kof  = (lane >> 4) * 8;
  bf16x16 b;
#pragma unroll
  for (int e = 0; e < 16; ++e) b[e] = p[col * ld + ((e < 8) ? e : (8 + e)) + kof];
  return b;
}

// ---------------------------------------------------------------------------
// Kernel 0: f32 -> bf16 weight conversion
// ---------------------------------------------------------------------------
__global__ void cvt_bf16(const float* __restrict__ src, __bf16* __restrict__ dst, int n) {
  int i = blockIdx.x * 256 + threadIdx.x;
  if (i < n) dst[i] = f2bf(src[i]);
}

// ---------------------------------------------------------------------------
// Kernel 1: projection GEMM  y = x @ W  (M=4096, N=512, K=512)
// block = 256 threads (8 waves); block tile 128x64, k-step 32.
// A tile: manual f32->bf16 staging.  B tile: TDM async load.
// Epilogue: optional relu+eps, bf16 store remapped to [b,h,n,dh].
// ---------------------------------------------------------------------------
__global__ __launch_bounds__(256) void proj_gemm(
    const float* __restrict__ X,   // [4096,512] f32
    const __bf16* __restrict__ W,  // [512,512] bf16 (k-major)
    __bf16* __restrict__ dst,      // [2][8][2048][64] bf16
    int applyRelu)
{
  __shared__ alignas(16) __bf16 As[128 * 32];
  __shared__ alignas(16) __bf16 Bs[32 * 64];
  int m0 = blockIdx.x * 128;
  int n0 = blockIdx.y * 64;
  int tid = threadIdx.x, wave = tid >> 5, lane = tid & 31;
  f32x8 acc[4] = {};
  for (int k0 = 0; k0 < 512; k0 += 32) {
    if (tid < 32) {   // wave 0: TDM load of B tile 32x64 (row stride 512)
      tdm_load_2d_bf16(lds_off(Bs), W + k0 * 512 + n0, 64, 32, 512);
    }
    for (int i = tid; i < 128 * 32; i += 256) {
      int r = i >> 5, c = i & 31;
      As[i] = f2bf(X[(m0 + r) * 512 + k0 + c]);
    }
    if (tid < 32) tdm_wait0();
    __syncthreads();
    bf16x16 a = ldsA(&As[(wave * 16) * 32], 32);
#pragma unroll
    for (int t = 0; t < 4; ++t) {
      bf16x16 b = ldsB(&Bs[t * 16], 64);
      acc[t] = wmma_bf16(a, b, acc[t]);
    }
    __syncthreads();
  }
#pragma unroll
  for (int t = 0; t < 4; ++t) {
#pragma unroll
    for (int r = 0; r < 8; ++r) {
      int m = m0 + wave * 16 + r + 8 * (lane >> 4);
      int c = n0 + t * 16 + (lane & 15);
      float v = acc[t][r];
      if (applyRelu) v = fmaxf(v, 0.0f) + EPS;
      int b = m >> 11, n = m & 2047;
      int h = c >> 6,  d = c & 63;
      dst[(((b * 8 + h) * 2048 + n) << 6) + d] = f2bf(v);
    }
  }
}

// ---------------------------------------------------------------------------
// Kernel 2: per-chunk stats  S_c = K_c^T V_c  [64d x 64e],  ksum_c[d]
// grid (32 chunks, 16 bh); 256 threads; K/V chunks staged via TDM.
// ---------------------------------------------------------------------------
__global__ __launch_bounds__(256) void chunk_stats(
    const __bf16* __restrict__ Kbuf,  // [16][2048][64]
    const __bf16* __restrict__ Vbuf,
    float* __restrict__ Sc,           // [16][32][64*64]
    float* __restrict__ ksc)          // [16][32][64]
{
  int ch = blockIdx.x, bh = blockIdx.y;
  const __bf16* Kp = Kbuf + (bh * 2048 + ch * 64) * 64;
  const __bf16* Vp = Vbuf + (bh * 2048 + ch * 64) * 64;
  __shared__ alignas(16) __bf16 Ks[64 * 64];
  __shared__ alignas(16) __bf16 Vs[64 * 64];
  int tid = threadIdx.x, wave = tid >> 5, lane = tid & 31;
  if (tid < 32) {
    tdm_load_2d_bf16(lds_off(Ks), Kp, 64, 64, 64);
    tdm_load_2d_bf16(lds_off(Vs), Vp, 64, 64, 64);
    tdm_wait0();
  }
  __syncthreads();
  float* Sout = Sc + (bh * 32 + ch) * 4096;
#pragma unroll
  for (int tt = 0; tt < 2; ++tt) {
    int t = wave * 2 + tt, td = t >> 2, te = t & 3;
    f32x8 acc = {};
#pragma unroll
    for (int nn = 0; nn < 2; ++nn) {                 // K-dim = n (64)
      bf16x16 a = ldsAT(&Ks[(nn * 32) * 64 + td * 16], 64);  // A[d][n]=K[n][d]
      bf16x16 b = ldsB (&Vs[(nn * 32) * 64 + te * 16], 64);  // V[n][e]
      acc = wmma_bf16(a, b, acc);
    }
#pragma unroll
    for (int r = 0; r < 8; ++r) {
      int d = td * 16 + r + 8 * (lane >> 4);
      int e = te * 16 + (lane & 15);
      Sout[d * 64 + e] = acc[r];
    }
  }
  if (tid < 64) {
    float s = 0.f;
    for (int n = 0; n < 64; ++n) s += bf2f(Ks[n * 64 + tid]);
    ksc[(bh * 32 + ch) * 64 + tid] = s;
  }
}

// ---------------------------------------------------------------------------
// Kernel 3: exclusive prefix scan over 32 chunks, element-parallel, in place.
// ---------------------------------------------------------------------------
__global__ __launch_bounds__(256) void chunk_scan(float* Sc, float* ksc) {
  int bh = blockIdx.x, tid = threadIdx.x;
  float* S = Sc + bh * 32 * 4096;
  for (int e = tid; e < 4096; e += 256) {
    float run = 0.f;
    for (int c = 0; c < 32; ++c) {
      float v = S[c * 4096 + e];
      S[c * 4096 + e] = run;
      run += v;
    }
  }
  if (tid < 64) {
    float* ks = ksc + bh * 32 * 64;
    float run = 0.f;
    for (int c = 0; c < 32; ++c) {
      float v = ks[c * 64 + tid];
      ks[c * 64 + tid] = run;
      run += v;
    }
  }
}

// ---------------------------------------------------------------------------
// Kernel 4: per-chunk output.
//   A = causal_mask(Q K^T); den[n] = Q.ksum_prev + rowsum(A);
//   out = (Q @ S_prev + A @ V) * (1/den)
// Q/K/V chunks staged via TDM; S_prev converted f32->bf16 by all threads.
// ---------------------------------------------------------------------------
__global__ __launch_bounds__(256) void attn_out(
    const __bf16* __restrict__ Qbuf, const __bf16* __restrict__ Kbuf,
    const __bf16* __restrict__ Vbuf,
    const float* __restrict__ Sc, const float* __restrict__ ksc,
    __bf16* __restrict__ Obuf)        // [2][2048][512]
{
  int ch = blockIdx.x, bh = blockIdx.y;
  int b = bh >> 3, h = bh & 7;
  const __bf16* Qp = Qbuf + (bh * 2048 + ch * 64) * 64;
  const __bf16* Kp = Kbuf + (bh * 2048 + ch * 64) * 64;
  const __bf16* Vp = Vbuf + (bh * 2048 + ch * 64) * 64;
  const float*  Sp = Sc  + (bh * 32 + ch) * 4096;
  const float*  kp = ksc + (bh * 32 + ch) * 64;
  __shared__ alignas(16) __bf16 Qs[64 * 64];
  __shared__ alignas(16) __bf16 Ks[64 * 64];
  __shared__ alignas(16) __bf16 Vs[64 * 64];
  __shared__ alignas(16) __bf16 As[64 * 64];   // masked Q K^T
  __shared__ alignas(16) __bf16 Sb[64 * 64];   // S_prev in bf16
  __shared__ float dinv[64];
  int tid = threadIdx.x, wave = tid >> 5, lane = tid & 31;
  if (tid < 32) {
    tdm_load_2d_bf16(lds_off(Qs), Qp, 64, 64, 64);
    tdm_load_2d_bf16(lds_off(Ks), Kp, 64, 64, 64);
    tdm_load_2d_bf16(lds_off(Vs), Vp, 64, 64, 64);
  }
  for (int i = tid; i < 4096; i += 256) Sb[i] = f2bf(Sp[i]);
  if (tid < 32) tdm_wait0();
  __syncthreads();
  // A = mask(Q K^T)
#pragma unroll
  for (int tt = 0; tt < 2; ++tt) {
    int t = wave * 2 + tt, tn = t >> 2, tm = t & 3;
    f32x8 acc = {};
#pragma unroll
    for (int dd = 0; dd < 2; ++dd) {
      bf16x16 a = ldsA (&Qs[(tn * 16) * 64 + dd * 32], 64);   // Q[n][d]
      bf16x16 b = ldsBT(&Ks[(tm * 16) * 64 + dd * 32], 64);   // B[d][m]=K[m][d]
      acc = wmma_bf16(a, b, acc);
    }
#pragma unroll
    for (int r = 0; r < 8; ++r) {
      int n = tn * 16 + r + 8 * (lane >> 4);
      int m = tm * 16 + (lane & 15);
      As[n * 64 + m] = f2bf((m <= n) ? acc[r] : 0.0f);
    }
  }
  __syncthreads();
  // denominator
  if (tid < 64) {
    int n = tid;
    float s = 0.f;
    for (int m = 0; m < 64; ++m) s += bf2f(As[n * 64 + m]);
    for (int d = 0; d < 64; ++d) s += bf2f(Qs[n * 64 + d]) * kp[d];
    dinv[n] = 1.0f / s;
  }
  __syncthreads();
  // out = (Q @ S_prev + A @ V) * dinv
  __bf16* Op = Obuf + ((long)b * 2048 + ch * 64) * 512 + h * 64;
#pragma unroll
  for (int tt = 0; tt < 2; ++tt) {
    int t = wave * 2 + tt, tn = t >> 2, te = t & 3;
    f32x8 acc = {};
#pragma unroll
    for (int dd = 0; dd < 2; ++dd) {
      bf16x16 a = ldsA(&Qs[(tn * 16) * 64 + dd * 32], 64);    // Q[n][d]
      bf16x16 b = ldsB(&Sb[(dd * 32) * 64 + te * 16], 64);    // S[d][e]
      acc = wmma_bf16(a, b, acc);
    }
#pragma unroll
    for (int mm = 0; mm < 2; ++mm) {
      bf16x16 a = ldsA(&As[(tn * 16) * 64 + mm * 32], 64);    // A[n][m]
      bf16x16 b = ldsB(&Vs[(mm * 32) * 64 + te * 16], 64);    // V[m][e]
      acc = wmma_bf16(a, b, acc);
    }
#pragma unroll
    for (int r = 0; r < 8; ++r) {
      int n = tn * 16 + r + 8 * (lane >> 4);
      int e = te * 16 + (lane & 15);
      Op[n * 512 + e] = f2bf(acc[r] * dinv[n]);
    }
  }
}

// ---------------------------------------------------------------------------
// Kernel 5: output projection  out = attn @ Wo + bo   (f32 result)
// Both A and B tiles staged via TDM (both already bf16).
// ---------------------------------------------------------------------------
__global__ __launch_bounds__(256) void out_gemm(
    const __bf16* __restrict__ A,   // [4096,512] bf16
    const __bf16* __restrict__ W,   // [512,512] bf16
    const float* __restrict__ bias, // [512]
    float* __restrict__ out)        // [4096,512] f32
{
  __shared__ alignas(16) __bf16 As[128 * 32];
  __shared__ alignas(16) __bf16 Bs[32 * 64];
  int m0 = blockIdx.x * 128;
  int n0 = blockIdx.y * 64;
  int tid = threadIdx.x, wave = tid >> 5, lane = tid & 31;
  f32x8 acc[4] = {};
  for (int k0 = 0; k0 < 512; k0 += 32) {
    if (tid < 32) {
      tdm_load_2d_bf16(lds_off(As), A + m0 * 512 + k0, 32, 128, 512);
      tdm_load_2d_bf16(lds_off(Bs), W + k0 * 512 + n0, 64, 32, 512);
      tdm_wait0();
    }
    __syncthreads();
    bf16x16 a = ldsA(&As[(wave * 16) * 32], 32);
#pragma unroll
    for (int t = 0; t < 4; ++t) {
      bf16x16 b = ldsB(&Bs[t * 16], 64);
      acc[t] = wmma_bf16(a, b, acc[t]);
    }
    __syncthreads();
  }
#pragma unroll
  for (int t = 0; t < 4; ++t) {
#pragma unroll
    for (int r = 0; r < 8; ++r) {
      int m = m0 + wave * 16 + r + 8 * (lane >> 4);
      int c = n0 + t * 16 + (lane & 15);
      out[m * 512 + c] = acc[t][r] + bias[c];
    }
  }
}

// ---------------------------------------------------------------------------
extern "C" void kernel_launch(void* const* d_in, const int* in_sizes, int n_in,
                              void* d_out, int out_size, void* d_ws, size_t ws_size,
                              hipStream_t stream) {
  (void)in_sizes; (void)n_in; (void)out_size; (void)ws_size;
  const float* x  = (const float*)d_in[0];
  const float* Wq = (const float*)d_in[1];
  const float* Wk = (const float*)d_in[2];
  const float* Wv = (const float*)d_in[3];
  const float* Wo = (const float*)d_in[4];
  const float* bo = (const float*)d_in[5];
  float* out = (float*)d_out;

  char* ws = (char*)d_ws;
  __bf16* wq_bf = (__bf16*)(ws + 0x0000000);  // 512KB each
  __bf16* wk_bf = (__bf16*)(ws + 0x0080000);
  __bf16* wv_bf = (__bf16*)(ws + 0x0100000);
  __bf16* wo_bf = (__bf16*)(ws + 0x0180000);
  __bf16* qbuf  = (__bf16*)(ws + 0x0200000); // [16][2048][64] bf16 = 4MB
  __bf16* kbuf  = (__bf16*)(ws + 0x0600000);
  __bf16* vbuf  = (__bf16*)(ws + 0x0A00000);
  __bf16* obuf  = (__bf16*)(ws + 0x0E00000); // [2][2048][512] bf16 = 4MB
  float*  Sc    = (float*) (ws + 0x1200000); // [16][32][4096] f32 = 8MB
  float*  ksc   = (float*) (ws + 0x1A00000); // [16][32][64]  f32

  const int WELEM = 512 * 512;
  cvt_bf16<<<WELEM / 256, 256, 0, stream>>>(Wq, wq_bf, WELEM);
  cvt_bf16<<<WELEM / 256, 256, 0, stream>>>(Wk, wk_bf, WELEM);
  cvt_bf16<<<WELEM / 256, 256, 0, stream>>>(Wv, wv_bf, WELEM);
  cvt_bf16<<<WELEM / 256, 256, 0, stream>>>(Wo, wo_bf, WELEM);

  dim3 gp(32, 8);
  proj_gemm<<<gp, 256, 0, stream>>>(x, wq_bf, qbuf, 1);
  proj_gemm<<<gp, 256, 0, stream>>>(x, wk_bf, kbuf, 1);
  proj_gemm<<<gp, 256, 0, stream>>>(x, wv_bf, vbuf, 0);

  chunk_stats<<<dim3(32, 16), 256, 0, stream>>>(kbuf, vbuf, Sc, ksc);
  chunk_scan<<<16, 256, 0, stream>>>(Sc, ksc);
  attn_out<<<dim3(32, 16), 256, 0, stream>>>(qbuf, kbuf, vbuf, Sc, ksc, obuf);

  out_gemm<<<gp, 256, 0, stream>>>(obuf, wo_bf, bo, out);
}
